// ConvolutionOneWay_72335839200038
// MI455X (gfx1250) — compile-verified, matrix operand order checked by reading, not run on hardware
//
#include <hip/hip_runtime.h>
#include <hip/hip_bf16.h>

typedef __attribute__((ext_vector_type(16))) _Float16 v16h;
typedef __attribute__((ext_vector_type(8)))  _Float16 v8h;
typedef __attribute__((ext_vector_type(8)))  float    v8f;
typedef __attribute__((ext_vector_type(4)))  float    f32x4;

#define M0c 64
#define M1c 32
#define DIMc 160   // 64 + 3*32
#define RFW 384    // rf0(96) + rf1(96*3)

// ---------------------------------------------------------------------------
// Kernel 1: node linear  out[n][0:64]   = (in0 @ W0) * attr / sqrt(64)
//                        out[n][64+3v+i]= sum_u in1[u][i] W1[u][v] * attr/sqrt(32)
// ---------------------------------------------------------------------------
__global__ void node_lin_kernel(const float* __restrict__ in,
                                const float* __restrict__ attr,
                                const float* __restrict__ W0,   // [64,64]
                                const float* __restrict__ W1,   // [32,32]
                                float* __restrict__ out, int N)
{
    long idx = (long)blockIdx.x * blockDim.x + threadIdx.x;
    long total = (long)N * DIMc;
    if (idx >= total) return;
    int n  = (int)(idx / DIMc);
    int ch = (int)(idx - (long)n * DIMc);
    const float* row = in + (long)n * DIMc;
    float a = attr[n];
    float s;
    if (ch < M0c) {
        float acc = 0.f;
        #pragma unroll 8
        for (int u = 0; u < M0c; ++u) acc += row[u] * W0[u * M0c + ch];
        s = acc * a * 0.125f;                       // 1/sqrt(64)
    } else {
        int v = (ch - M0c) / 3, i = (ch - M0c) % 3;
        float acc = 0.f;
        #pragma unroll 8
        for (int u = 0; u < M1c; ++u) acc += row[M0c + 3 * u + i] * W1[u * M1c + v];
        s = acc * a * 0.17677669529663689f;         // 1/sqrt(32)
    }
    out[idx] = s;
}

// ---------------------------------------------------------------------------
// Kernel 2: zero the receiver accumulator
// ---------------------------------------------------------------------------
__global__ void zero_kernel(float* __restrict__ p, long n)
{
    long i = (long)blockIdx.x * blockDim.x + threadIdx.x;
    if (i < n) p[i] = 0.f;
}

// ---------------------------------------------------------------------------
// Kernel 3: fused edge MLP (WMMA f16->f32) + tensor product + atomic scatter
// Block: 64 threads (2 waves), 32 edges per block, 16 edges per wave.
// LDS weights are stored pre-swizzled in the WMMA B lane layout so each
// B fragment is a single contiguous 32B read per lane.
// ---------------------------------------------------------------------------
__global__ __launch_bounds__(64) void edge_mlp_tp_kernel(
    const float* __restrict__ edge_attr,     // [E,4]
    const float* __restrict__ edge_scalars,  // [E,16]
    const float* __restrict__ fc_W1,         // [16,64]
    const float* __restrict__ fc_W2,         // [64,192]
    const int*   __restrict__ edge_src,
    const int*   __restrict__ edge_dst,
    const float* __restrict__ f_s,           // [NS,160] sender features
    float*       __restrict__ rf,            // [NR,384] accumulator
    int E)
{
    // w1swz[t][lane][e], e=0..7 (K-pattern per lane), scaled by 1/4
    __shared__ __attribute__((aligned(16))) _Float16 w1swz[4 * 32 * 8];
    // w2swz[kc][t][lane][e], e=0..15, scaled by 1/8
    __shared__ __attribute__((aligned(32))) _Float16 w2swz[2 * 12 * 32 * 16];
    // silu(h) per wave, row-major [row][K] so GEMM2-A reads are contiguous
    __shared__ __attribute__((aligned(16))) _Float16 hbuf[2][16 * 64];
    // per-edge weights, transposed [col][row] so D-spill is vectorized
    __shared__ __attribute__((aligned(16))) float wt_t[2][192 * 16];
    __shared__ float s_y[32][4];
    __shared__ int   s_src[32], s_dst[32];

    const int tid  = threadIdx.x;
    const int wave = tid >> 5;
    const int lane = tid & 31;
    const int m    = lane & 15;   // A row / B col / D col within tile
    const int half = lane >> 4;   // K-octet / D row-group select
    const int blockEdge0 = blockIdx.x * 32;

    // ---- Preload + swizzle MLP weights into WMMA lane layout (f16) ----
    for (int i = tid; i < 16 * 64; i += 64) {
        int K = i >> 6, n = i & 63;
        int h = K >> 3, e = K & 7;                  // K<16
        int ln = h * 16 + (n & 15), t = n >> 4;
        w1swz[(t * 32 + ln) * 8 + e] = (_Float16)(fc_W1[i] * 0.25f);
    }
    for (int i = tid; i < 64 * 192; i += 64) {
        int K = i / 192, n = i - K * 192;
        int kc = K >> 5, Kr = K & 31;
        int h, e;
        if (Kr < 16) { h = Kr >> 3; e = Kr & 7; }
        else         { h = (Kr - 16) >> 3; e = 8 + ((Kr - 16) & 7); }
        int ln = h * 16 + (n & 15), t = n >> 4;
        w2swz[(((kc * 12 + t) * 32 + ln) << 4) + e] = (_Float16)(fc_W2[i] * 0.125f);
    }
    // ---- Stage per-edge metadata ----
    for (int i = tid; i < 32; i += 64) {
        int ge = blockEdge0 + i;
        if (ge < E) { s_src[i] = edge_src[ge]; s_dst[i] = edge_dst[ge]; }
        else        { s_src[i] = 0;            s_dst[i] = 0; }
    }
    for (int i = tid; i < 128; i += 64) {
        int ge = blockEdge0 + (i >> 2);
        s_y[i >> 2][i & 3] = (ge < E) ? edge_attr[(long)ge * 4 + (i & 3)] : 0.f;
    }
    __syncthreads();

    // Prefetch gathered sender rows (L2-resident) behind the WMMA MLP.
    for (int i = tid; i < 64; i += 64 /* one iter */) {}
    for (int i = tid; i < 64; i += 64) {
        int e = i >> 1;
        const float* p = f_s + (long)s_src[e] * DIMc + (i & 1) * 80;
        __builtin_prefetch(p, 0, 0);
    }

    const int tileEdge0 = blockEdge0 + wave * 16;

    // ---- GEMM1: h[16x64] = silu( es[16x16] @ W1s[16x64] ), K padded 16->32 ----
    v16h a;
    {
        int ed = tileEdge0 + m;
        if (ed < E) {
            const f32x4* rp = (const f32x4*)(edge_scalars + (long)ed * 16 + half * 8);
            f32x4 r0 = rp[0], r1 = rp[1];
            a[0] = (_Float16)r0.x; a[1] = (_Float16)r0.y;
            a[2] = (_Float16)r0.z; a[3] = (_Float16)r0.w;
            a[4] = (_Float16)r1.x; a[5] = (_Float16)r1.y;
            a[6] = (_Float16)r1.z; a[7] = (_Float16)r1.w;
        } else {
            #pragma unroll
            for (int e = 0; e < 8; ++e) a[e] = (_Float16)0.f;
        }
        #pragma unroll
        for (int e = 8; e < 16; ++e) a[e] = (_Float16)0.f;  // K>=16 pad
    }
    #pragma unroll
    for (int t = 0; t < 4; ++t) {
        v8h blo = *(const v8h*)&w1swz[(t * 32 + lane) * 8];
        v16h b;
        #pragma unroll
        for (int e = 0; e < 8; ++e) { b[e] = blo[e]; b[e + 8] = (_Float16)0.f; }
        v8f c = {};
        c = __builtin_amdgcn_wmma_f32_16x16x32_f16(false, a, false, b,
                                                   (short)0, c, false, false);
        #pragma unroll
        for (int r = 0; r < 8; ++r) {
            float x = c[r];
            float s = x * __builtin_amdgcn_rcpf(1.0f + __expf(-x));  // silu
            hbuf[wave][(r + half * 8) * 64 + t * 16 + m] = (_Float16)s;
        }
    }
    __syncthreads();

    // ---- GEMM2: weight[16x192] = h[16x64] @ W2s[64x192] ----
    v8f acc[12] = {};
    #pragma unroll
    for (int kc = 0; kc < 2; ++kc) {
        v8h alo = *(const v8h*)&hbuf[wave][m * 64 + kc * 32 + half * 8];
        v8h ahi = *(const v8h*)&hbuf[wave][m * 64 + kc * 32 + 16 + half * 8];
        v16h a2;
        #pragma unroll
        for (int e = 0; e < 8; ++e) { a2[e] = alo[e]; a2[e + 8] = ahi[e]; }
        #pragma unroll
        for (int t = 0; t < 12; ++t) {
            v16h b2 = *(const v16h*)&w2swz[(((kc * 12 + t) * 32 + lane) << 4)];
            acc[t] = __builtin_amdgcn_wmma_f32_16x16x32_f16(false, a2, false, b2,
                                                            (short)0, acc[t],
                                                            false, false);
        }
    }
    // D layout: VGPR r holds row (r + 8*half), column (16t + m).
    // Spill transposed: wt_t[col*16 + row], 8 contiguous floats per (t, lane).
    #pragma unroll
    for (int t = 0; t < 12; ++t) {
        int base = (16 * t + m) * 16 + 8 * half;
        f32x4 lo, hi;
        lo.x = acc[t][0]; lo.y = acc[t][1]; lo.z = acc[t][2]; lo.w = acc[t][3];
        hi.x = acc[t][4]; hi.y = acc[t][5]; hi.z = acc[t][6]; hi.w = acc[t][7];
        *(f32x4*)&wt_t[wave][base]     = lo;
        *(f32x4*)&wt_t[wave][base + 4] = hi;
    }
    __syncthreads();

    // ---- Tensor product + atomic scatter (all 64 lanes over 32 edges) ----
    const float inv3 = 0.57735026918962576f;   // 1/sqrt(3)
    for (int i = tid; i < 32 * 96; i += 64) {
        int e = i / 96;
        int u = i - e * 96;
        int ge = blockEdge0 + e;
        if (ge >= E) continue;
        int src = s_src[e], dst = s_dst[e];
        float y0  = s_y[e][0];
        float y1x = s_y[e][1], y1y = s_y[e][2], y1z = s_y[e][3];
        const float* w    = &wt_t[e >> 4][e & 15];          // + col*16
        const float* xrow = f_s + (long)src * DIMc;
        float* r0 = rf + (long)dst * RFW;
        float* r1 = r0 + 96;
        if (u < 64) {
            float x0 = xrow[u];
            float wa = w[u * 16], wb = w[(64 + u) * 16];
            atomicAdd(&r0[u], wa * x0 * y0);                     // out_a
            float bx = wb * x0;                                  // out_b
            atomicAdd(&r1[3 * u + 0], bx * y1x);
            atomicAdd(&r1[3 * u + 1], bx * y1y);
            atomicAdd(&r1[3 * u + 2], bx * y1z);
        } else {
            int v = u - 64;
            float xx = xrow[M0c + 3 * v + 0];
            float xy = xrow[M0c + 3 * v + 1];
            float xz = xrow[M0c + 3 * v + 2];
            float wc = w[(128 + v) * 16], wd = w[(160 + v) * 16];
            atomicAdd(&r0[u], wd * (xx * y1x + xy * y1y + xz * y1z) * inv3); // out_d
            atomicAdd(&r1[3 * u + 0], wc * xx * y0);             // out_c
            atomicAdd(&r1[3 * u + 1], wc * xy * y0);
            atomicAdd(&r1[3 * u + 2], wc * xz * y0);
        }
    }
}

// ---------------------------------------------------------------------------
// Kernel 4: receiver finalize  (conv linears + rotation gate)
// ---------------------------------------------------------------------------
__global__ void finalize_kernel(const float* __restrict__ rf,   // [NR,384]
                                const float* __restrict__ sc,   // [NR,160]
                                const float* __restrict__ rattr,
                                const float* __restrict__ W20,  // [96,64]
                                const float* __restrict__ W21,  // [96,32]
                                const float* __restrict__ W3,   // [96]
                                float* __restrict__ out, int NR)
{
    long idx = (long)blockIdx.x * blockDim.x + threadIdx.x;
    long total = (long)NR * DIMc;
    if (idx >= total) return;
    int n  = (int)(idx / DIMc);
    int ch = (int)(idx - (long)n * DIMc);
    const float* r = rf + (long)n * RFW;
    float a = rattr[n];
    const float scale = a * 0.10206207261596575f * 0.25f; // (1/sqrt(96)) * (1/sqrt(16))

    float ang = 0.f;
    #pragma unroll 8
    for (int u = 0; u < 96; ++u) ang += r[u] * W3[u];
    ang *= 0.1f * scale;
    float cs = cosf(ang), sn = sinf(ang);

    float conv;
    if (ch < M0c) {
        float acc = 0.f;
        #pragma unroll 8
        for (int u = 0; u < 96; ++u) acc += r[u] * W20[u * M0c + ch];
        conv = acc * scale;
    } else {
        int v = (ch - M0c) / 3, i = (ch - M0c) % 3;
        float acc = 0.f;
        #pragma unroll 8
        for (int u = 0; u < 96; ++u) acc += r[96 + 3 * u + i] * W21[u * M1c + v];
        conv = acc * scale;
    }
    out[idx] = cs * sc[idx] + sn * conv;
}

// ---------------------------------------------------------------------------
extern "C" void kernel_launch(void* const* d_in, const int* in_sizes, int n_in,
                              void* d_out, int out_size, void* d_ws, size_t ws_size,
                              hipStream_t stream)
{
    const float* sender_input   = (const float*)d_in[0];
    const float* sender_attr    = (const float*)d_in[1];
    const float* receiver_input = (const float*)d_in[2];
    const float* receiver_attr  = (const float*)d_in[3];
    const float* edge_attr      = (const float*)d_in[4];
    const float* edge_scalars   = (const float*)d_in[5];
    const float* W_sc0          = (const float*)d_in[6];
    const float* W_sc1          = (const float*)d_in[7];
    const float* W_lin1_0       = (const float*)d_in[8];
    const float* W_lin1_1       = (const float*)d_in[9];
    const float* fc_W1          = (const float*)d_in[10];
    const float* fc_W2          = (const float*)d_in[11];
    const float* W_lin2_0       = (const float*)d_in[12];
    const float* W_lin2_1       = (const float*)d_in[13];
    const float* W_lin3         = (const float*)d_in[14];
    const int*   edge_src       = (const int*)d_in[15];
    const int*   edge_dst       = (const int*)d_in[16];
    float* out = (float*)d_out;

    const int NS = in_sizes[0] / DIMc;
    const int NR = in_sizes[2] / DIMc;
    const int E  = in_sizes[15];

    // Workspace layout (floats): f_s[NS*160] | sc[NR*160] | rf[NR*384]
    float* f_s = (float*)d_ws;
    float* sc  = f_s + (size_t)NS * DIMc;
    float* rf  = sc  + (size_t)NR * DIMc;

    // 1) node linears (sender features + receiver self-connection)
    {
        long t1 = (long)NS * DIMc;
        node_lin_kernel<<<(int)((t1 + 255) / 256), 256, 0, stream>>>(
            sender_input, sender_attr, W_lin1_0, W_lin1_1, f_s, NS);
        long t2 = (long)NR * DIMc;
        node_lin_kernel<<<(int)((t2 + 255) / 256), 256, 0, stream>>>(
            receiver_input, receiver_attr, W_sc0, W_sc1, sc, NR);
    }
    // 2) clear accumulator
    {
        long nrf = (long)NR * RFW;
        zero_kernel<<<(int)((nrf + 255) / 256), 256, 0, stream>>>(rf, nrf);
    }
    // 3) fused edge MLP (WMMA) + tensor product + atomic scatter
    {
        int blocks = (E + 31) / 32;
        edge_mlp_tp_kernel<<<blocks, 64, 0, stream>>>(
            edge_attr, edge_scalars, fc_W1, fc_W2, edge_src, edge_dst, f_s, rf, E);
    }
    // 4) finalize
    {
        long t = (long)NR * DIMc;
        finalize_kernel<<<(int)((t + 255) / 256), 256, 0, stream>>>(
            rf, sc, receiver_attr, W_lin2_0, W_lin2_1, W_lin3, out, NR);
    }
}
